// ConditionalGraphNoisePred_14250701488267
// MI455X (gfx1250) — matrix operand order repaired and checked by reading, not compiled, not run on hardware
//
#include <hip/hip_runtime.h>

// ---------------------------------------------------------------------------
// EGNN diffusion forward on gfx1250 (MI455X).
// Bandwidth-bound (~400 GFLOP vs several GB of HBM traffic): bf16 WMMA GEMMs
// (v_wmma_f32_16x16x32_bf16), weights pre-transposed/converted to bf16 and
// staged into LDS with global_load_async_to_lds_b64 (ASYNCcnt), fused A-side
// gathers, rank-1 edge embedding, f32 global atomics for segment sums.
// ---------------------------------------------------------------------------

constexpr int N_NODES  = 8192;
constexpr int N_EDGES  = 65536;
constexpr int E_TOT    = N_EDGES + N_NODES;   // 73728 (with self loops)
constexpr int N_GRAPHS = 64;
constexpr int NODE_F   = 8;
constexpr int PRED_H   = 16;
constexpr int HID      = 256;
constexpr int EMB      = 32;
constexpr int DD       = HID + EMB;           // 288
constexpr int L_MAIN   = 5;
constexpr int L_COND   = 3;
constexpr int COND_CH  = DD * 2 * L_MAIN;     // 2880

typedef __bf16 bf16;
typedef __attribute__((ext_vector_type(16))) __bf16 v16bf;
typedef __attribute__((ext_vector_type(8)))  __bf16 v8bf;
typedef __attribute__((ext_vector_type(8)))  float  v8f;

#define DEV static __device__ __forceinline__

DEV bf16 f2bf(float f) { return (bf16)f; }   // native cvt (RNE) on gfx1250

DEV int e_row(const int* ei, int e) { return (e < N_EDGES) ? ei[e] : (e - N_EDGES); }
DEV int e_col(const int* ei, int e) { return (e < N_EDGES) ? ei[N_EDGES + e] : (e - N_EDGES); }

DEV void ld8(const float* __restrict__ p, float* o) {
  float4 a = *(const float4*)p;
  float4 b = *(const float4*)(p + 4);
  o[0] = a.x; o[1] = a.y; o[2] = a.z; o[3] = a.w;
  o[4] = b.x; o[5] = b.y; o[6] = b.z; o[7] = b.w;
}

DEV void wait_async0() {
#if __has_builtin(__builtin_amdgcn_s_wait_asynccnt)
  __builtin_amdgcn_s_wait_asynccnt(0);
#else
  asm volatile("s_wait_asynccnt 0" ::: "memory");
#endif
}

// ---------------------------------------------------------------------------
// Generic WMMA GEMM:  C[M x N] = act( gatherA[M x K] @ W[K x N] + bias )
// Block tile 64x32, 8 waves (4 M x 2 N), each wave one 16x16 WMMA tile, K=32.
// Weights arrive pre-transposed (N-major) in bf16, zero-padded to Kpad.
// ---------------------------------------------------------------------------
enum { GA_PLAIN = 0, GA_EDGE1 = 1, GA_NODE1 = 2 };
enum { ACT_NONE = 0, ACT_SILU = 1, ACT_RELU = 2 };

struct GemmGather {
  const float* hsrc;      // node features (stride nf)
  int          nf;
  const float* magg;      // aggregated messages (stride HID)
  const float* radial;    // per-edge radial
  const float* edge_attr; // raw edge_attr (N_EDGES entries; self loops -> 0)
  const int*   eidx;      // edge_index (2 x N_EDGES)
  const float* heW;       // edge_emb rank-1 weight (HID) or null
  const float* heB;       // edge_emb bias (HID) or null
};

// Load one 8-wide A chunk. pr/pc are precomputed per-row segment base ptrs.
// Segment boundaries (nf, 2nf, nf+HID) are 8-aligned -> chunks are single
// contiguous segments except the radial/rank-1 boundary and the K tail.
template <int MODE>
DEV void chunk8(const float* __restrict__ pr, const float* __restrict__ pc,
                float eaA, float radA, int nf, const float* __restrict__ heW,
                const float* __restrict__ heB, int k0, int Kact, float* o) {
  if (MODE == GA_PLAIN) {
    if (k0 + 8 <= Kact) { ld8(pr + k0, o); return; }
#pragma unroll
    for (int i = 0; i < 8; ++i) { int k = k0 + i; o[i] = (k < Kact) ? pr[k] : 0.0f; }
    return;
  }
  if (MODE == GA_NODE1) {
    if (k0 + 8 <= nf) ld8(pr + k0, o);
    else              ld8(pc + (k0 - nf), o);
    return;
  }
  // GA_EDGE1: [h[row] (nf) | h[col] (nf) | radial (1) | edge_attr (ecols)]
  if (k0 + 8 <= nf) { ld8(pr + k0, o); return; }
  if (k0 >= nf && k0 + 8 <= 2 * nf) { ld8(pc + (k0 - nf), o); return; }
  if (heW && k0 > 2 * nf && k0 + 8 <= Kact) {      // rank-1 edge-emb region
    int j = k0 - 2 * nf - 1;
#pragma unroll
    for (int i = 0; i < 8; ++i) o[i] = eaA * heW[j + i] + heB[j + i];
    return;
  }
#pragma unroll
  for (int i = 0; i < 8; ++i) {                    // boundary / tail chunk
    int k = k0 + i;
    float v = 0.0f;
    if (k < nf)           v = pr[k];
    else if (k < 2 * nf)  v = pc[k - nf];
    else if (k == 2 * nf) v = radA;
    else if (k < Kact)    v = heW ? (eaA * heW[k - 2 * nf - 1] + heB[k - 2 * nf - 1]) : eaA;
    o[i] = v;
  }
}

#define BM 64
#define BN 32
#define BK 32
#define APAD 40   // padded row stride (80B, keeps 16B alignment)

template <int MODE, int ACT, bool ACCUM>
__global__ __launch_bounds__(256)
void k_gemm(const float* __restrict__ A, int lda,
            const bf16* __restrict__ Wt, int Kact,
            const float* __restrict__ bias, float* __restrict__ C, int ldc,
            GemmGather ga) {
  __shared__ __align__(16) bf16 As[BM][APAD];
  __shared__ __align__(16) bf16 Bs[BN][APAD];   // N-major: Bs[n][k]

  const int t    = threadIdx.x;
  const int lane = t & 31;
  const int wave = t >> 5;
  const int wm   = wave & 3;        // 4 M sub-tiles
  const int wn   = wave >> 2;       // 2 N sub-tiles
  const int m0   = blockIdx.x * BM;
  const int n0   = blockIdx.y * BN;
  const int Kpad = (Kact + 31) & ~31;

  // ---- per-thread A staging context (hoisted out of K loop)
  const int sr  = t >> 2;            // staging row 0..63
  const int skk = (t & 3) * 8;       // staging k offset within tile
  const int g_r = m0 + sr;
  const float* pr = nullptr;
  const float* pc = nullptr;
  float eaA = 0.0f, radA = 0.0f;
  if (MODE == GA_PLAIN) {
    pr = A + (size_t)g_r * lda;
  } else if (MODE == GA_EDGE1) {
    pr   = ga.hsrc + (size_t)e_row(ga.eidx, g_r) * ga.nf;
    pc   = ga.hsrc + (size_t)e_col(ga.eidx, g_r) * ga.nf;
    eaA  = (g_r < N_EDGES) ? ga.edge_attr[g_r] : 0.0f;
    radA = ga.radial[g_r];
  } else {
    pr = ga.hsrc + (size_t)g_r * ga.nf;
    pc = ga.magg + (size_t)g_r * HID;
  }

  // ---- per-thread B staging: async-copy 8 bytes (4 bf16) of one column
  const int bn  = t >> 3;            // local column 0..31
  const int bko = (t & 7) * 4;       // local k offset 0..28
  const bf16* wsrc = Wt + (size_t)(n0 + bn) * Kpad + bko;
  // LDS aperture flat addresses carry the LDS byte offset in the low 32 bits
  const unsigned bdst = (unsigned)(unsigned long long)&Bs[bn][bko];

  v8f acc = {0.f, 0.f, 0.f, 0.f, 0.f, 0.f, 0.f, 0.f};

  for (int kt = 0; kt < Kpad; kt += BK) {
    // ---- stage B: async global->LDS copy (tracked by ASYNCcnt)
    {
      const bf16* gsrc = wsrc + kt;
      asm volatile("global_load_async_to_lds_b64 %0, %1, off"
                   :: "v"(bdst), "v"(gsrc) : "memory");
    }
    // ---- stage A: 8-wide gathered chunk -> pack bf16 -> one 16B LDS store
    {
      float o[8];
      chunk8<MODE>(pr, pc, eaA, radA, ga.nf, ga.heW, ga.heB, kt + skk, Kact, o);
      v8bf pk;
#pragma unroll
      for (int i = 0; i < 8; ++i) pk[i] = f2bf(o[i]);
      *(v8bf*)&As[sr][skk] = pk;
    }
    wait_async0();
    __syncthreads();

    // ---- WMMA fragments per CDNA5 wave32 layout (CDNA5 ISA)
    const int half = lane >> 4, l15 = lane & 15;
    // A: lanes 0-15 hold K{0..7,16..23} of row M=l15; lanes 16-31 K{8..15,24..31}
    v8bf alo = *(const v8bf*)&As[wm * 16 + l15][half * 8];
    v8bf ahi = *(const v8bf*)&As[wm * 16 + l15][16 + half * 8];
    // B: lanes 0-15 hold K{0..7,16..23} of column N=l15; lanes 16-31 K{8..15,24..31}
    v8bf blo = *(const v8bf*)&Bs[wn * 16 + l15][half * 8];
    v8bf bhi = *(const v8bf*)&Bs[wn * 16 + l15][16 + half * 8];
    v16bf av, bv;
#pragma unroll
    for (int i = 0; i < 8; ++i) {
      av[i] = alo[i]; av[8 + i] = ahi[i];
      bv[i] = blo[i]; bv[8 + i] = bhi[i];
    }
    acc = __builtin_amdgcn_wmma_f32_16x16x32_bf16(
        false, av, false, bv, (short)0, acc, false, false);
    __syncthreads();
  }

  // ---- store C: VGPR v -> row M = half*8+v, col N = l15
  const int l15  = lane & 15;
  const int half = lane >> 4;
  const int col  = n0 + wn * 16 + l15;
  const float bb = bias[col];
#pragma unroll
  for (int v = 0; v < 8; ++v) {
    int    row = m0 + wm * 16 + half * 8 + v;
    float  val = acc[v] + bb;
    if (ACT == ACT_SILU)      val = val / (1.0f + __expf(-val));
    else if (ACT == ACT_RELU) val = fmaxf(val, 0.0f);
    size_t idx = (size_t)row * ldc + col;
    if (ACCUM) val += C[idx];
    C[idx] = val;
  }
}

// ---------------------------------------------------------------------------
// Weight prep: W (K x N, f32, row-major) -> Wt (N x Kpad, bf16, zero-padded)
// ---------------------------------------------------------------------------
__global__ void k_wprep(const float* __restrict__ W, bf16* __restrict__ Wt,
                        int K, int N, int Kpad) {
  int id = blockIdx.x * blockDim.x + threadIdx.x;   // N*Kpad
  int n = id / Kpad, k = id % Kpad;
  Wt[(size_t)n * Kpad + k] = f2bf((k < K) ? W[(size_t)k * N + n] : 0.0f);
}

// ---------------------------------------------------------------------------
// Small kernels
// ---------------------------------------------------------------------------

// posenc -> dse1 -> mish -> dse2, one block per graph
__global__ void k_time_embed(const int* __restrict__ ts,
                             const float* __restrict__ w1, const float* __restrict__ b1,
                             const float* __restrict__ w2, const float* __restrict__ b2,
                             float* __restrict__ te) {
  __shared__ float pe[EMB];
  __shared__ float hid[4 * EMB];
  int g = blockIdx.x, t = threadIdx.x;
  if (t < EMB) {
    float pos  = (float)ts[g] + 1.0f;
    int   pair = t >> 1;
    float div  = __expf((float)(2 * pair) * (-logf(10000.0f) / (float)EMB));
    float ang  = pos * div;
    pe[t] = (t & 1) ? cosf(ang) : sinf(ang);
  }
  __syncthreads();
  if (t < 4 * EMB) {
    float s = b1[t];
    for (int i = 0; i < EMB; ++i) s += pe[i] * w1[i * (4 * EMB) + t];
    float sp = logf(1.0f + __expf(s));           // softplus
    hid[t] = s * tanhf(sp);                      // mish
  }
  __syncthreads();
  if (t < EMB) {
    float s = b2[t];
    for (int i = 0; i < 4 * EMB; ++i) s += hid[i] * w2[i * EMB + t];
    te[g * EMB + t] = s;
  }
}

__global__ void k_set_te(float* __restrict__ h, const float* __restrict__ te,
                         const int* __restrict__ batch) {
  int id = blockIdx.x * blockDim.x + threadIdx.x;  // N_NODES*EMB
  int n = id >> 5, j = id & 31;
  h[(size_t)n * DD + HID + j] = te[batch[n] * EMB + j];
}

__global__ void k_edge_geom(const float* __restrict__ coord, const int* __restrict__ ei,
                            int normalize, float* __restrict__ cdiff,
                            float* __restrict__ radial) {
  int e = blockIdx.x * blockDim.x + threadIdx.x;
  if (e >= E_TOT) return;
  int r = e_row(ei, e), c = e_col(ei, e);
  float dx = coord[r * 3 + 0] - coord[c * 3 + 0];
  float dy = coord[r * 3 + 1] - coord[c * 3 + 1];
  float dz = coord[r * 3 + 2] - coord[c * 3 + 2];
  float rad = dx * dx + dy * dy + dz * dz;
  radial[e] = rad;
  if (normalize) {
    float inv = 1.0f / (sqrtf(rad) + 1e-8f);
    dx *= inv; dy *= inv; dz *= inv;
  }
  cdiff[e * 3 + 0] = dx; cdiff[e * 3 + 1] = dy; cdiff[e * 3 + 2] = dz;
}

// t[e] = dot(silu(coord1(m))[e], coord2_w) ; one wave per edge
__global__ void k_coord2(const float* __restrict__ msrc, const float* __restrict__ w,
                         float* __restrict__ tout) {
  int wid  = (blockIdx.x * blockDim.x + threadIdx.x) >> 5;
  int lane = threadIdx.x & 31;
  if (wid >= E_TOT) return;
  const float* mr = msrc + (size_t)wid * HID;
  float s = 0.0f;
  for (int i = lane; i < HID; i += 32) s += mr[i] * w[i];
  for (int off = 16; off; off >>= 1) s += __shfl_down(s, off);
  if (lane == 0) tout[wid] = s;
}

__global__ void k_coord_scatter(const float* __restrict__ cdiff,
                                const float* __restrict__ tsc,
                                const int* __restrict__ ei,
                                float* __restrict__ cagg, float* __restrict__ cnt) {
  int e = blockIdx.x * blockDim.x + threadIdx.x;
  if (e >= E_TOT) return;
  int r = e_row(ei, e);
  float tv = tsc[e];
  atomicAdd(&cagg[r * 3 + 0], cdiff[e * 3 + 0] * tv);
  atomicAdd(&cagg[r * 3 + 1], cdiff[e * 3 + 1] * tv);
  atomicAdd(&cagg[r * 3 + 2], cdiff[e * 3 + 2] * tv);
  atomicAdd(&cnt[r], 1.0f);
}

__global__ void k_coord_update(float* __restrict__ coord, const float* __restrict__ cagg,
                               const float* __restrict__ cnt) {
  int n = blockIdx.x * blockDim.x + threadIdx.x;
  if (n >= N_NODES) return;
  float c = fmaxf(cnt[n], 1.0f);
  coord[n * 3 + 0] += cagg[n * 3 + 0] / c;
  coord[n * 3 + 1] += cagg[n * 3 + 1] / c;
  coord[n * 3 + 2] += cagg[n * 3 + 2] / c;
}

__global__ void k_magg_scatter(const float* __restrict__ m, const int* __restrict__ ei,
                               float* __restrict__ magg) {
  int id = blockIdx.x * blockDim.x + threadIdx.x;  // E_TOT*HID
  int e = id >> 8, ch = id & 255;
  int r = e_row(ei, e);
  atomicAdd(&magg[(size_t)r * HID + ch], m[(size_t)e * HID + ch]);
}

__global__ void k_pool_scatter(const float* __restrict__ src, const int* __restrict__ batch,
                               float* __restrict__ g, float* __restrict__ gcnt) {
  int id = blockIdx.x * blockDim.x + threadIdx.x;  // N_NODES*HID
  int n = id >> 8, ch = id & 255;
  int b = batch[n];
  atomicAdd(&g[(size_t)b * HID + ch], src[(size_t)n * HID + ch]);
  if (ch == 0) atomicAdd(&gcnt[b], 1.0f);
}

__global__ void k_pool_div(float* __restrict__ g, const float* __restrict__ gcnt) {
  int id = blockIdx.x * blockDim.x + threadIdx.x;  // N_GRAPHS*HID
  g[id] /= gcnt[id >> 8];
}

// h = scales[l][batch]*h + biases[l][batch]  (embed flat: ((l*64+g)*2+s)*288+c)
__global__ void k_film(float* __restrict__ h, const float* __restrict__ embed,
                       const int* __restrict__ batch, int l) {
  int id = blockIdx.x * blockDim.x + threadIdx.x;  // N_NODES*DD
  int n = id / DD, c = id % DD;
  int b = batch[n];
  size_t base = ((size_t)(l * N_GRAPHS + b) * 2) * DD;
  h[id] = embed[base + c] * h[id] + embed[base + DD + c];
}

// ---------------------------------------------------------------------------
// Host side
// ---------------------------------------------------------------------------
struct EgclP {
  const float *c1b, *c1w, *c2w, *e1b, *e1w, *e2b, *e2w, *n1b, *n1w, *n2b, *n2w;
};
struct EgclT {   // bf16 pre-transposed weights
  const bf16 *e1t, *e2t, *c1t, *n1t, *n2t;
};

static char* carveB(char*& p, size_t bytes) {
  char* r = p;
  p += ((bytes + 255) & ~(size_t)255);
  return r;
}
static float* carve(char*& p, size_t nf) { return (float*)carveB(p, nf * sizeof(float)); }

static void gemm(hipStream_t s, int mode, const float* A, int lda, const bf16* Wt,
                 int Kact, int M, int N, const float* bias, float* C, int ldc,
                 int act, int accum, const GemmGather& ga) {
  dim3 grid(M / BM, N / BN);
#define KG(MO, AC, ACC) \
  k_gemm<MO, AC, ACC><<<grid, 256, 0, s>>>(A, lda, Wt, Kact, bias, C, ldc, ga)
  if (mode == GA_EDGE1) { KG(GA_EDGE1, ACT_SILU, false); return; }
  if (mode == GA_NODE1) { KG(GA_NODE1, ACT_SILU, false); return; }
  if (accum)            { KG(GA_PLAIN, ACT_NONE, true);  return; }
  if (act == ACT_SILU)  { KG(GA_PLAIN, ACT_SILU, false); return; }
  if (act == ACT_RELU)  { KG(GA_PLAIN, ACT_RELU, false); return; }
  KG(GA_PLAIN, ACT_NONE, false);
#undef KG
}

extern "C" void kernel_launch(void* const* d_in, const int* in_sizes, int n_in,
                              void* d_out, int out_size, void* d_ws, size_t ws_size,
                              hipStream_t stream) {
  (void)in_sizes; (void)n_in; (void)out_size; (void)ws_size;
  auto F = [&](int i) { return (const float*)d_in[i]; };
  auto I = [&](int i) { return (const int*)d_in[i]; };

  // ---- base inputs (setup_inputs order)
  const float* x       = F(0);   // (8192,16,8)
  const int*   eidx    = I(1);   // (2,65536)
  const float* eattr   = F(2);   // (65536,)
  const float* x_coord = F(3);   // (8192,3)
  const float* cond    = F(4);   // (8192,2,4)
  const int*   tsteps  = I(5);   // (64,)
  const int*   batch   = I(6);   // (8192,)

  // ---- params: jax tree-flatten (dicts sorted by key, lists in order)
  const float* ce_in_b  = F(7);
  const float* ce_in_w  = F(8);
  const float* ce_out_b = F(9);
  const float* ce_out_w = F(10);
  const float* ce_fc_b  = F(11);
  const float* ce_fc_w  = F(12);
  auto egclAt = [&](int base) {
    return EgclP{F(base), F(base + 1), F(base + 2), F(base + 3), F(base + 4),
                 F(base + 5), F(base + 6), F(base + 7), F(base + 8),
                 F(base + 9), F(base + 10)};
  };
  const float* dse1_b = F(46);
  const float* dse1_w = F(47);
  const float* dse2_b = F(48);
  const float* dse2_w = F(49);
  const float* ew_b   = F(50);   // edge_emb bias (256)
  const float* ew_w   = F(51);   // edge_emb weight (1,256)
  const float* ne_b = F(107);
  const float* ne_w = F(108);
  const float* p0_b = F(109), *p0_w = F(110);
  const float* p1_b = F(111), *p1_w = F(112);
  const float* p2_b = F(113), *p2_w = F(114);

  // ---- workspace carve
  char* wp = (char*)d_ws;
  float* h      = carve(wp, (size_t)N_NODES * DD);
  float* hc     = carve(wp, (size_t)N_NODES * HID);
  float* ntmp   = carve(wp, (size_t)N_NODES * HID);
  float* magg   = carve(wp, (size_t)N_NODES * HID);
  float* eb0    = carve(wp, (size_t)E_TOT * HID);
  float* eb1    = carve(wp, (size_t)E_TOT * HID);
  float* xc     = carve(wp, (size_t)N_NODES * 3);
  float* xv     = carve(wp, (size_t)N_NODES * 3);
  float* cagg   = carve(wp, (size_t)N_NODES * 3);
  float* cnt    = carve(wp, N_NODES);
  float* radial = carve(wp, E_TOT);
  float* cdiff  = carve(wp, (size_t)E_TOT * 3);
  float* tsc    = carve(wp, E_TOT);
  float* te     = carve(wp, N_GRAPHS * EMB);
  float* gbuf   = carve(wp, N_GRAPHS * HID);
  float* gcnt   = carve(wp, N_GRAPHS);
  float* embed  = carve(wp, (size_t)N_GRAPHS * COND_CH);

  // ---- bf16 weight arena: transpose+convert every GEMM weight once per call
  auto prep = [&](const float* W, int K, int N) -> const bf16* {
    int Kpad = (K + 31) & ~31;
    bf16* dst = (bf16*)carveB(wp, (size_t)N * Kpad * sizeof(bf16));
    k_wprep<<<((size_t)N * Kpad) / 256, 256, 0, stream>>>(W, dst, K, N, Kpad);
    return dst;
  };
  const bf16* ne_t   = prep(ne_w, PRED_H * NODE_F, HID);
  const bf16* cein_t = prep(ce_in_w, 4 * 2, HID);
  const bf16* ceout_t= prep(ce_out_w, HID, HID);
  const bf16* fc_t   = prep(ce_fc_w, HID, COND_CH);
  const bf16* p0_t   = prep(p0_w, DD, HID);
  const bf16* p1_t   = prep(p1_w, HID, HID);
  const bf16* p2_t   = prep(p2_w, HID, PRED_H * NODE_F);
  EgclP cP[L_COND], mP[L_MAIN];
  EgclT cT[L_COND], mT[L_MAIN];
  for (int i = 0; i < L_COND; ++i) {
    cP[i] = egclAt(13 + 11 * i);
    cT[i] = EgclT{prep(cP[i].e1w, 2 * HID + 2, HID), prep(cP[i].e2w, HID, HID),
                  prep(cP[i].c1w, HID, HID),         prep(cP[i].n1w, 2 * HID, HID),
                  prep(cP[i].n2w, HID, HID)};
  }
  for (int i = 0; i < L_MAIN; ++i) {
    mP[i] = egclAt(52 + 11 * i);
    mT[i] = EgclT{prep(mP[i].e1w, 2 * DD + 1 + HID, HID), prep(mP[i].e2w, HID, HID),
                  prep(mP[i].c1w, HID, HID),              prep(mP[i].n1w, DD + HID, HID),
                  prep(mP[i].n2w, HID, DD)};
  }

  const GemmGather g0{};

  // one EGCL layer (on features `hfeat` of width nf and coords `coord`)
  auto run_egcl = [&](const EgclP& p, const EgclT& w, float* hfeat, int nf,
                      float* coord, int normalize, bool isMain) {
    k_edge_geom<<<E_TOT / 256, 256, 0, stream>>>(coord, eidx, normalize, cdiff, radial);
    GemmGather gaE{hfeat, nf, nullptr, radial, eattr, eidx,
                   isMain ? ew_w : nullptr, isMain ? ew_b : nullptr};
    int Ke = isMain ? (2 * DD + 1 + HID) : (2 * HID + 2);   // 833 / 514
    gemm(stream, GA_EDGE1, nullptr, 0, w.e1t, Ke, E_TOT, HID, p.e1b, eb0, HID, ACT_SILU, 0, gaE);
    gemm(stream, GA_PLAIN, eb0, HID, w.e2t, HID, E_TOT, HID, p.e2b, eb1, HID, ACT_SILU, 0, g0);
    // coord path: eb1 = m
    gemm(stream, GA_PLAIN, eb1, HID, w.c1t, HID, E_TOT, HID, p.c1b, eb0, HID, ACT_SILU, 0, g0);
    k_coord2<<<E_TOT / 8, 256, 0, stream>>>(eb0, p.c2w, tsc);
    hipMemsetAsync(cagg, 0, (size_t)N_NODES * 3 * sizeof(float), stream);
    hipMemsetAsync(cnt, 0, (size_t)N_NODES * sizeof(float), stream);
    k_coord_scatter<<<E_TOT / 256, 256, 0, stream>>>(cdiff, tsc, eidx, cagg, cnt);
    k_coord_update<<<N_NODES / 256, 256, 0, stream>>>(coord, cagg, cnt);
    // node path
    hipMemsetAsync(magg, 0, (size_t)N_NODES * HID * sizeof(float), stream);
    k_magg_scatter<<<(E_TOT * HID) / 256, 256, 0, stream>>>(eb1, eidx, magg);
    GemmGather gaN{hfeat, nf, magg, nullptr, nullptr, nullptr, nullptr, nullptr};
    gemm(stream, GA_NODE1, nullptr, 0, w.n1t, nf + HID, N_NODES, HID, p.n1b, ntmp, HID, ACT_SILU, 0, gaN);
    gemm(stream, GA_PLAIN, ntmp, HID, w.n2t, HID, N_NODES, nf, p.n2b, hfeat, nf, ACT_NONE, 1, g0);
  };

  // ---- time embedding, h = [node_emb(x), te[batch]]
  k_time_embed<<<N_GRAPHS, 128, 0, stream>>>(tsteps, dse1_w, dse1_b, dse2_w, dse2_b, te);
  gemm(stream, GA_PLAIN, x, PRED_H * NODE_F, ne_t, PRED_H * NODE_F, N_NODES, HID,
       ne_b, h, DD, ACT_NONE, 0, g0);
  k_set_te<<<(N_NODES * EMB) / 256, 256, 0, stream>>>(h, te, batch);

  // ---- conditioning path
  gemm(stream, GA_PLAIN, cond, 4 * 2, cein_t, 4 * 2, N_NODES, HID, ce_in_b, hc, HID,
       ACT_NONE, 0, g0);
  hipMemcpyAsync(xc, x_coord, (size_t)N_NODES * 3 * sizeof(float),
                 hipMemcpyDeviceToDevice, stream);
  hipMemcpyAsync(xv, x_coord, (size_t)N_NODES * 3 * sizeof(float),
                 hipMemcpyDeviceToDevice, stream);
  for (int cl = 0; cl < L_COND; ++cl)
    run_egcl(cP[cl], cT[cl], hc, HID, xc, /*normalize=*/0, /*isMain=*/false);

  gemm(stream, GA_PLAIN, hc, HID, ceout_t, HID, N_NODES, HID, ce_out_b, ntmp, HID,
       ACT_NONE, 0, g0);
  hipMemsetAsync(gbuf, 0, (size_t)N_GRAPHS * HID * sizeof(float), stream);
  hipMemsetAsync(gcnt, 0, (size_t)N_GRAPHS * sizeof(float), stream);
  k_pool_scatter<<<(N_NODES * HID) / 256, 256, 0, stream>>>(ntmp, batch, gbuf, gcnt);
  k_pool_div<<<(N_GRAPHS * HID) / 256, 256, 0, stream>>>(gbuf, gcnt);
  gemm(stream, GA_PLAIN, gbuf, HID, fc_t, HID, N_GRAPHS, COND_CH, ce_fc_b, embed,
       COND_CH, ACT_NONE, 0, g0);

  // ---- main EGNN layers with FiLM conditioning
  for (int l = 0; l < L_MAIN; ++l) {
    k_film<<<(N_NODES * DD) / 256, 256, 0, stream>>>(h, embed, batch, l);
    run_egcl(mP[l], mT[l], h, DD, xv, /*normalize=*/1, /*isMain=*/true);
  }

  // ---- prediction head -> d_out[0 : N*128], then coords -> d_out[N*128 : ]
  float* dout = (float*)d_out;
  gemm(stream, GA_PLAIN, h, DD, p0_t, DD, N_NODES, HID, p0_b, hc, HID, ACT_RELU, 0, g0);
  gemm(stream, GA_PLAIN, hc, HID, p1_t, HID, N_NODES, HID, p1_b, ntmp, HID, ACT_RELU, 0, g0);
  gemm(stream, GA_PLAIN, ntmp, HID, p2_t, HID, N_NODES, PRED_H * NODE_F, p2_b, dout,
       PRED_H * NODE_F, ACT_NONE, 0, g0);
  hipMemcpyAsync(dout + (size_t)N_NODES * PRED_H * NODE_F, xv,
                 (size_t)N_NODES * 3 * sizeof(float), hipMemcpyDeviceToDevice, stream);
}
